// CrossAttention_11828339933333
// MI455X (gfx1250) — compile-verified, matrix operand order checked by reading, not run on hardware
//
#include <hip/hip_runtime.h>
#include <hip/hip_bf16.h>

// ---------------------------------------------------------------------------
// CDNA5 (gfx1250) fused cross-attention, bf16 WMMA pipeline, 2x2 reg blocking,
// software-pipelined k-loop (unroll 2 to kill rotation moves),
// uniform-branch epilogues.
// Shapes: B=4, Qn=900, S=5, N=3600 (60x60), SN=18000, C=256.
// ---------------------------------------------------------------------------

typedef unsigned short u16;
typedef __attribute__((ext_vector_type(16))) u16      v16u;
typedef __attribute__((ext_vector_type(8)))  u16      v8u;
typedef __attribute__((ext_vector_type(8)))  unsigned v8i;
typedef __attribute__((ext_vector_type(16))) __bf16   v16bf;
typedef __attribute__((ext_vector_type(8)))  float    v8f;
typedef __attribute__((ext_vector_type(4)))  float    v4f;

#define NB 4
#define QN 900
#define SN 18000
#define CC 256
#define HH 60

// f32 -> bf16 round-to-nearest-even (scalar)
__device__ __forceinline__ u16 f2bf(float f) {
    unsigned u = __float_as_uint(f);
    u += 0x7fffu + ((u >> 16) & 1u);
    return (u16)(u >> 16);
}
// packed pair: a -> low 16, b -> high 16
__device__ __forceinline__ unsigned f2bf_pk(float a, float b) {
#if __has_builtin(__builtin_amdgcn_cvt_pk_bf16_f32)
    auto t = __builtin_amdgcn_cvt_pk_bf16_f32(a, b);
    return __builtin_bit_cast(unsigned, t);
#else
    return (unsigned)f2bf(a) | ((unsigned)f2bf(b) << 16);
#endif
}

// ---------------------------------------------------------------------------
// Fragment loaders for V_WMMA_F32_16X16X32_BF16 (ISA 7.12.2):
// lanes 0-15 : row = lane,    K = {0..7, 16..23}
// lanes 16-31: row = lane-16, K = {8..15, 24..31}
// rowptr points at (row, k0); all addresses 16B-aligned by construction.
// ---------------------------------------------------------------------------
__device__ __forceinline__ v16u load_frag_full(const u16* __restrict__ p, int half) {
    const int o0 = half * 8, o1 = 16 + half * 8;
    v8u lo = *(const v8u*)(p + o0);
    v8u hi = *(const v8u*)(p + o1);
    v16u r;
#pragma unroll
    for (int j = 0; j < 8; ++j) { r[j] = lo[j]; r[8 + j] = hi[j]; }
    return r;
}

__device__ __forceinline__ v16u load_frag_full(const float* __restrict__ p, int half) {
    const int o0 = half * 8, o1 = 16 + half * 8;
    v4f f0 = *(const v4f*)(p + o0);
    v4f f1 = *(const v4f*)(p + o0 + 4);
    v4f f2 = *(const v4f*)(p + o1);
    v4f f3 = *(const v4f*)(p + o1 + 4);
    v8i pk;
    pk[0] = f2bf_pk(f0[0], f0[1]); pk[1] = f2bf_pk(f0[2], f0[3]);
    pk[2] = f2bf_pk(f1[0], f1[1]); pk[3] = f2bf_pk(f1[2], f1[3]);
    pk[4] = f2bf_pk(f2[0], f2[1]); pk[5] = f2bf_pk(f2[2], f2[3]);
    pk[6] = f2bf_pk(f3[0], f3[1]); pk[7] = f2bf_pk(f3[2], f3[3]);
    return __builtin_bit_cast(v16u, pk);
}

// 16-wide K tail (krem == 16): run0 (K = half*8 .. half*8+7) fully in range,
// run1 (K >= 16) fully out of range -> zeros.
__device__ __forceinline__ v16u load_frag_tail16(const u16* __restrict__ p, int half) {
    v8u lo = *(const v8u*)(p + half * 8);
    v16u r;
#pragma unroll
    for (int j = 0; j < 8; ++j) { r[j] = lo[j]; r[8 + j] = (u16)0; }
    return r;
}
__device__ __forceinline__ v16u load_frag_tail16(const float* __restrict__ p, int half) {
    v4f f0 = *(const v4f*)(p + half * 8);
    v4f f1 = *(const v4f*)(p + half * 8 + 4);
    v8i pk;
    pk[0] = f2bf_pk(f0[0], f0[1]); pk[1] = f2bf_pk(f0[2], f0[3]);
    pk[2] = f2bf_pk(f1[0], f1[1]); pk[3] = f2bf_pk(f1[2], f1[3]);
    pk[4] = 0u; pk[5] = 0u; pk[6] = 0u; pk[7] = 0u;
    return __builtin_bit_cast(v16u, pk);
}

__device__ __forceinline__ v8f wmma_bf16(v16u a, v16u b, v8f c) {
    return __builtin_amdgcn_wmma_f32_16x16x32_bf16(
        false, __builtin_bit_cast(v16bf, a),
        false, __builtin_bit_cast(v16bf, b),
        (short)0, c, false, false);
}

// 2x2 tile (32M x 32N) GEMM core, software-pipelined: next k-step's fragment
// loads are issued before the current 4 WMMAs so VMEM overlaps the XDL pipe.
// Unroll 2 so register allocation alternates fragment tuples (no rotate moves).
template <typename TA, typename TB>
__device__ __forceinline__ void gemm_2x2(const TA* __restrict__ a0,
                                         const TA* __restrict__ a1,
                                         const TB* __restrict__ b0,
                                         const TB* __restrict__ b1,
                                         int K, int half, v8f* acc) {
    v16u A0 = load_frag_full(a0, half);
    v16u A1 = load_frag_full(a1, half);
    v16u B0 = load_frag_full(b0, half);
    v16u B1 = load_frag_full(b1, half);
    int k0 = 32;
#pragma unroll 2
    for (; k0 + 32 <= K; k0 += 32) {
        __builtin_prefetch(a0 + k0 + 32, 0, 1);   // global_prefetch_b8
        __builtin_prefetch(b0 + k0 + 32, 0, 1);
        v16u nA0 = load_frag_full(a0 + k0, half);
        v16u nA1 = load_frag_full(a1 + k0, half);
        v16u nB0 = load_frag_full(b0 + k0, half);
        v16u nB1 = load_frag_full(b1 + k0, half);
        acc[0] = wmma_bf16(A0, B0, acc[0]);
        acc[1] = wmma_bf16(A0, B1, acc[1]);
        acc[2] = wmma_bf16(A1, B0, acc[2]);
        acc[3] = wmma_bf16(A1, B1, acc[3]);
        A0 = nA0; A1 = nA1; B0 = nB0; B1 = nB1;
    }
    acc[0] = wmma_bf16(A0, B0, acc[0]);
    acc[1] = wmma_bf16(A0, B1, acc[1]);
    acc[2] = wmma_bf16(A1, B0, acc[2]);
    acc[3] = wmma_bf16(A1, B1, acc[3]);
    if (k0 < K) {  // exactly 16 remaining (AV GEMM: 18000 = 562*32 + 16)
        v16u tA0 = load_frag_tail16(a0 + k0, half);
        v16u tA1 = load_frag_tail16(a1 + k0, half);
        v16u tB0 = load_frag_tail16(b0 + k0, half);
        v16u tB1 = load_frag_tail16(b1 + k0, half);
        acc[0] = wmma_bf16(tA0, tB0, acc[0]);
        acc[1] = wmma_bf16(tA0, tB1, acc[1]);
        acc[2] = wmma_bf16(tA1, tB0, acc[2]);
        acc[3] = wmma_bf16(tA1, tB1, acc[3]);
    }
}

__device__ __forceinline__ int clampi(int x, int hi) { return x > hi ? hi : x; }

// ---------------------------------------------------------------------------
// K1: Y[M,256](bf16) = X[M,256](f32) @ W[256,256]^T(f32)
// grid(x = 256/32 = 8, y = ceil(M/32)), block 32
// ---------------------------------------------------------------------------
__global__ __launch_bounds__(32) void proj_gemm_kernel(
    const float* __restrict__ X, const float* __restrict__ W,
    u16* __restrict__ Y, int M) {
    const int lane = threadIdx.x;
    const int half = lane >> 4, lr = lane & 15;
    const int n0 = blockIdx.x * 32;
    const int m0 = blockIdx.y * 32;
    const float* a0 = X + (size_t)clampi(m0 + lr,      M - 1) * CC;
    const float* a1 = X + (size_t)clampi(m0 + 16 + lr, M - 1) * CC;
    const float* b0 = W + (size_t)(n0 + lr) * CC;
    const float* b1 = W + (size_t)(n0 + 16 + lr) * CC;
    v8f acc[4] = {};
    gemm_2x2(a0, a1, b0, b1, CC, half, acc);
    if (m0 + 31 < M) {  // uniform: straight-line stores
#pragma unroll
        for (int i = 0; i < 2; ++i)
#pragma unroll
            for (int j = 0; j < 2; ++j)
#pragma unroll
                for (int r = 0; r < 8; ++r) {
                    int m = m0 + 16 * i + half * 8 + r;
                    int n = n0 + 16 * j + lr;
                    Y[(size_t)m * CC + n] = f2bf(acc[i * 2 + j][r]);
                }
    } else {
#pragma unroll
        for (int i = 0; i < 2; ++i)
#pragma unroll
            for (int j = 0; j < 2; ++j)
#pragma unroll
                for (int r = 0; r < 8; ++r) {
                    int m = m0 + 16 * i + half * 8 + r;
                    int n = n0 + 16 * j + lr;
                    if (m < M) Y[(size_t)m * CC + n] = f2bf(acc[i * 2 + j][r]);
                }
    }
}

// ---------------------------------------------------------------------------
// K2: logits[b,m,n] = scale * <qf[b,m,:], kf[b,n,:]> + maskbias
// grid(x = ceil(SN/32) = 563, y = ceil(QN/32) = 29, z = B), block 32
// ---------------------------------------------------------------------------
__global__ __launch_bounds__(32) void attn_gemm_kernel(
    const u16* __restrict__ qf, const u16* __restrict__ kf,
    const unsigned char* __restrict__ mask,  // jax bool: 1 byte/element
    float* __restrict__ logits) {
    const int lane = threadIdx.x;
    const int half = lane >> 4, lr = lane & 15;
    const int n0 = blockIdx.x * 32;
    const int m0 = blockIdx.y * 32;
    const int b  = blockIdx.z;
    const u16* a0 = qf + ((size_t)b * QN + clampi(m0 + lr,      QN - 1)) * CC;
    const u16* a1 = qf + ((size_t)b * QN + clampi(m0 + 16 + lr, QN - 1)) * CC;
    const u16* b0 = kf + ((size_t)b * SN + clampi(n0 + lr,      SN - 1)) * CC;
    const u16* b1 = kf + ((size_t)b * SN + clampi(n0 + 16 + lr, SN - 1)) * CC;
    v8f acc[4] = {};
    gemm_2x2(a0, a1, b0, b1, CC, half, acc);

    const float scale = 0.0625f;  // 256^-0.5
    // mask bias depends only on n (invariant over the lane's 16 m values)
    float biasj[2];
#pragma unroll
    for (int j = 0; j < 2; ++j) {
        int n = clampi(n0 + 16 * j + lr, SN - 1);
        biasj[j] = mask[(size_t)b * SN + n] ? 0.0f : -10000.0f;
    }
    float* Lb = logits + (size_t)b * QN * SN;
    if (m0 + 31 < QN && n0 + 31 < SN) {  // uniform: straight-line stores
#pragma unroll
        for (int i = 0; i < 2; ++i)
#pragma unroll
            for (int j = 0; j < 2; ++j)
#pragma unroll
                for (int r = 0; r < 8; ++r) {
                    int m = m0 + 16 * i + half * 8 + r;
                    int n = n0 + 16 * j + lr;
                    Lb[(size_t)m * SN + n] = acc[i * 2 + j][r] * scale + biasj[j];
                }
    } else {
#pragma unroll
        for (int i = 0; i < 2; ++i)
#pragma unroll
            for (int j = 0; j < 2; ++j)
#pragma unroll
                for (int r = 0; r < 8; ++r) {
                    int m = m0 + 16 * i + half * 8 + r;
                    int n = n0 + 16 * j + lr;
                    if (m < QN && n < SN)
                        Lb[(size_t)m * SN + n] = acc[i * 2 + j][r] * scale + biasj[j];
                }
    }
}

// ---------------------------------------------------------------------------
// K3: per-row (b,m): 5x(60x60) gaussian blur + residual, softmax over 18000,
// bf16 probabilities written IN-PLACE over the row's own f32 region
// (row byte stride stays 72000B -> no cross-row aliasing).
// Whole 18000-elem row lives in LDS (72KB) -- CDNA5's 320KB WGP LDS.
// ---------------------------------------------------------------------------
__global__ __launch_bounds__(256) void blur_softmax_kernel(
    float* __restrict__ logits, const float* __restrict__ gauss) {
    extern __shared__ float sm[];
    float* row = sm;              // 18000
    float* tmp = sm + SN;         // 3600
    float* red = sm + SN + 3600;  // 256
    const int tid = threadIdx.x;
    const size_t r = blockIdx.x;
    float* L = logits + r * SN;

    float g[25];
#pragma unroll
    for (int i = 0; i < 25; ++i) g[i] = gauss[i];

    for (int s = 0; s < 5; ++s) {
        for (int i = tid; i < 3600; i += 256) tmp[i] = L[s * 3600 + i];
        __syncthreads();
        for (int i = tid; i < 3600; i += 256) {
            int y = i / HH, x = i % HH;
            float acc = 0.0f;
#pragma unroll
            for (int dy = -2; dy <= 2; ++dy) {
                int yy = y + dy;
                if (yy < 0 || yy >= HH) continue;
#pragma unroll
                for (int dx = -2; dx <= 2; ++dx) {
                    int xx = x + dx;
                    if (xx < 0 || xx >= HH) continue;
                    acc += g[(dy + 2) * 5 + (dx + 2)] * tmp[yy * HH + xx];
                }
            }
            row[s * 3600 + i] = tmp[i] + acc;  // residual add
        }
        __syncthreads();
    }

    float lm = -3.4e38f;
    for (int i = tid; i < SN; i += 256) lm = fmaxf(lm, row[i]);
    red[tid] = lm; __syncthreads();
    for (int off = 128; off > 0; off >>= 1) {
        if (tid < off) red[tid] = fmaxf(red[tid], red[tid + off]);
        __syncthreads();
    }
    const float rmax = red[0];
    __syncthreads();

    float ls = 0.0f;
    for (int i = tid; i < SN; i += 256) {
        float e = __expf(row[i] - rmax);
        row[i] = e;
        ls += e;
    }
    red[tid] = ls; __syncthreads();
    for (int off = 128; off > 0; off >>= 1) {
        if (tid < off) red[tid] += red[tid + off];
        __syncthreads();
    }
    const float inv = 1.0f / red[0];

    u16* P = (u16*)L;
    for (int i = tid; i < SN; i += 256) P[i] = f2bf(row[i] * inv);
}

// ---------------------------------------------------------------------------
// K4: vfT[b,c,k] = vf[b,k,c]
// ---------------------------------------------------------------------------
__global__ __launch_bounds__(256) void transpose_vf_kernel(
    const u16* __restrict__ vf, u16* __restrict__ vfT) {
    __shared__ u16 t[16][17];
    const int b = blockIdx.z;
    const int k0 = blockIdx.x * 16, c0 = blockIdx.y * 16;
    const int tx = threadIdx.x, ty = threadIdx.y;
    t[ty][tx] = vf[((size_t)b * SN + k0 + ty) * CC + c0 + tx];
    __syncthreads();
    vfT[((size_t)b * CC + c0 + ty) * SN + k0 + tx] = t[tx][ty];
}

// ---------------------------------------------------------------------------
// K5: x1[b,m,c] = sum_n P[b,m,n] * vfT[b,c,n]   (K = 18000, 16-wide tail)
// P rows: bf16 aliased over logits rows, row stride 72000 bytes.
// grid(256/32 = 8, 29, B), block 32.
// ---------------------------------------------------------------------------
__global__ __launch_bounds__(32) void av_gemm_kernel(
    const void* __restrict__ Pbase, const u16* __restrict__ vfT,
    u16* __restrict__ x1) {
    const int lane = threadIdx.x;
    const int half = lane >> 4, lr = lane & 15;
    const int n0 = blockIdx.x * 32;
    const int m0 = blockIdx.y * 32;
    const int b  = blockIdx.z;
    const u16* a0 = (const u16*)((const char*)Pbase +
        ((size_t)b * QN + clampi(m0 + lr,      QN - 1)) * (size_t)(SN * 4));
    const u16* a1 = (const u16*)((const char*)Pbase +
        ((size_t)b * QN + clampi(m0 + 16 + lr, QN - 1)) * (size_t)(SN * 4));
    const u16* b0 = vfT + ((size_t)b * CC + n0 + lr) * SN;
    const u16* b1 = vfT + ((size_t)b * CC + n0 + 16 + lr) * SN;
    v8f acc[4] = {};
    gemm_2x2(a0, a1, b0, b1, SN, half, acc);
    u16* Xb = x1 + (size_t)b * QN * CC;
    if (m0 + 31 < QN) {
#pragma unroll
        for (int i = 0; i < 2; ++i)
#pragma unroll
            for (int j = 0; j < 2; ++j)
#pragma unroll
                for (int r = 0; r < 8; ++r) {
                    int m = m0 + 16 * i + half * 8 + r;
                    int n = n0 + 16 * j + lr;
                    Xb[(size_t)m * CC + n] = f2bf(acc[i * 2 + j][r]);
                }
    } else {
#pragma unroll
        for (int i = 0; i < 2; ++i)
#pragma unroll
            for (int j = 0; j < 2; ++j)
#pragma unroll
                for (int r = 0; r < 8; ++r) {
                    int m = m0 + 16 * i + half * 8 + r;
                    int n = n0 + 16 * j + lr;
                    if (m < QN) Xb[(size_t)m * CC + n] = f2bf(acc[i * 2 + j][r]);
                }
    }
}

// ---------------------------------------------------------------------------
// K6: out[m,n](f32) = sum_k x1[m,k](bf16) * Wproj[n,k](f32)
// grid(8, ceil(3600/32) = 113), block 32
// ---------------------------------------------------------------------------
__global__ __launch_bounds__(32) void out_gemm_kernel(
    const u16* __restrict__ x1, const float* __restrict__ Wp,
    float* __restrict__ out) {
    const int lane = threadIdx.x;
    const int half = lane >> 4, lr = lane & 15;
    const int n0 = blockIdx.x * 32;
    const int m0 = blockIdx.y * 32;
    const int M = NB * QN;  // 3600
    const u16*   a0 = x1 + (size_t)clampi(m0 + lr,      M - 1) * CC;
    const u16*   a1 = x1 + (size_t)clampi(m0 + 16 + lr, M - 1) * CC;
    const float* b0 = Wp + (size_t)(n0 + lr) * CC;
    const float* b1 = Wp + (size_t)(n0 + 16 + lr) * CC;
    v8f acc[4] = {};
    gemm_2x2(a0, a1, b0, b1, CC, half, acc);
    if (m0 + 31 < M) {
#pragma unroll
        for (int i = 0; i < 2; ++i)
#pragma unroll
            for (int j = 0; j < 2; ++j)
#pragma unroll
                for (int r = 0; r < 8; ++r) {
                    int m = m0 + 16 * i + half * 8 + r;
                    int n = n0 + 16 * j + lr;
                    out[(size_t)m * CC + n] = acc[i * 2 + j][r];
                }
    } else {
#pragma unroll
        for (int i = 0; i < 2; ++i)
#pragma unroll
            for (int j = 0; j < 2; ++j)
#pragma unroll
                for (int r = 0; r < 8; ++r) {
                    int m = m0 + 16 * i + half * 8 + r;
                    int n = n0 + 16 * j + lr;
                    if (m < M) out[(size_t)m * CC + n] = acc[i * 2 + j][r];
                }
    }
}

// ---------------------------------------------------------------------------
extern "C" void kernel_launch(void* const* d_in, const int* in_sizes, int n_in,
                              void* d_out, int out_size, void* d_ws,
                              size_t ws_size, hipStream_t stream) {
    const float* q     = (const float*)d_in[0];         // (4,1,900,256)
    const float* k     = (const float*)d_in[1];         // (4,5,3600,256)
    const float* v     = (const float*)d_in[2];         // (4,5,3600,256)
    const unsigned char* mask = (const unsigned char*)d_in[3];  // (4,5,3600) bool
    const float* Wq    = (const float*)d_in[4];
    const float* Wk    = (const float*)d_in[5];
    const float* Wv    = (const float*)d_in[6];
    const float* Wproj = (const float*)d_in[7];
    const float* gauss = (const float*)d_in[8];         // (5,5)
    float* out = (float*)d_out;

    char* w = (char*)d_ws;
    const size_t SZ_QF  = (size_t)NB * QN * CC * 2;   //  1.84 MB bf16
    const size_t SZ_KF  = (size_t)NB * SN * CC * 2;   // 36.86 MB bf16
    u16*   QF  = (u16*)(w);
    u16*   KF  = (u16*)(w + SZ_QF);
    u16*   VF  = (u16*)(w + SZ_QF + SZ_KF);
    u16*   VFT = (u16*)(w + SZ_QF + 2 * SZ_KF);
    float* LOG = (float*)(w + SZ_QF + 3 * SZ_KF);     // 259.2 MB
    u16*   X1  = (u16*)((char*)LOG + (size_t)NB * QN * SN * 4);

    proj_gemm_kernel<<<dim3(8, 113),  32, 0, stream>>>(q, Wq, QF, NB * QN);
    proj_gemm_kernel<<<dim3(8, 2250), 32, 0, stream>>>(k, Wk, KF, NB * SN);
    proj_gemm_kernel<<<dim3(8, 2250), 32, 0, stream>>>(v, Wv, VF, NB * SN);

    transpose_vf_kernel<<<dim3(1125, 16, NB), dim3(16, 16), 0, stream>>>(VF, VFT);

    attn_gemm_kernel<<<dim3(563, 29, NB), 32, 0, stream>>>(QF, KF, mask, LOG);

    const size_t shmem = (size_t)(SN + 3600 + 256) * 4;  // 87,424 B
    blur_softmax_kernel<<<NB * QN, 256, shmem, stream>>>(LOG, gauss);

    av_gemm_kernel<<<dim3(8, 29, NB), 32, 0, stream>>>(LOG, VFT, X1);

    out_gemm_kernel<<<dim3(8, 113), 32, 0, stream>>>(X1, Wproj, out);
}
